// PointerDecoderSort_481036337478
// MI455X (gfx1250) — compile-verified, matrix operand order checked by reading, not run on hardware
//
#include <hip/hip_runtime.h>
#include <hip/hip_fp16.h>

typedef __attribute__((ext_vector_type(16))) _Float16 v16h;
typedef __attribute__((ext_vector_type(8)))  _Float16 v8h;
typedef __attribute__((ext_vector_type(8)))  float    v8f;

#define TILE_M 128
#define TILE_N 128
#define TILE_K 32
#define NSEQ 100
#define HDIM 64

// ---------------------------------------------------------------------------
// WMMA f16 GEMM:  C[m,n] = epi( sum_k A[m,k] * W[n,k] )
//   A: f16 [M,K] row-major,  W: f16 [N,K] row-major (B = W^T)
// Block tile 128x128, 8 waves, each wave 32(M) x 64(N): 8 WMMA / K-step.
// Global->LDS via async-to-LDS (ASYNCcnt) with double-buffered tiles;
// fragments load as 2x ds_load_b128 (ISA 16x32 f16 layout).
// Epilogue: +bias[n], +addend[m,n], +posadd[m%100,n], optional relu,
//           dual f32/f16 outputs. N multiple of 128 required; M guarded
//           (out-of-range rows clamp on load, skip on store).
// ---------------------------------------------------------------------------
__global__ __launch_bounds__(256)
void gemm_wmma_f16(const _Float16* __restrict__ A,
                   const _Float16* __restrict__ W,
                   const float* __restrict__ bias,
                   const float* __restrict__ addend,
                   const float* __restrict__ posadd,
                   float* __restrict__ outF,
                   _Float16* __restrict__ outH,
                   int M, int N, int K, int relu)
{
    __shared__ _Float16 As[2][TILE_M][TILE_K];   // 2 x 8 KB
    __shared__ _Float16 Bs[2][TILE_N][TILE_K];   // 2 x 8 KB

    const int tid  = threadIdx.x;
    const int wave = tid >> 5;      // 0..7
    const int lane = tid & 31;
    const int wm   = wave & 3;      // 32-row group
    const int wn   = wave >> 2;     // 64-col group
    const int rowBase = blockIdx.y * TILE_M;
    const int colBase = blockIdx.x * TILE_N;
    const int lhalf = lane >> 4;    // 0/1
    const int l15   = lane & 15;

    // per-thread cooperative-load assignment: 4 x 16B chunks per K-step
    const int cr = tid >> 2;              // 0..63
    const int cc = (tid & 3) * 8;         // 0,8,16,24 (halves)
    int agr0 = rowBase + cr;       if (agr0 >= M) agr0 = M - 1;   // clamp: rows >= M
    int agr1 = rowBase + cr + 64;  if (agr1 >= M) agr1 = M - 1;   // are never stored
    const _Float16* ap0 = A + (size_t)agr0 * K + cc;
    const _Float16* ap1 = A + (size_t)agr1 * K + cc;
    const _Float16* bp0 = W + (size_t)(colBase + cr) * K + cc;
    const _Float16* bp1 = W + (size_t)(colBase + cr + 64) * K + cc;
    const unsigned la0 = (unsigned)(uintptr_t)&As[0][cr][cc];
    const unsigned la1 = (unsigned)(uintptr_t)&As[0][cr + 64][cc];
    const unsigned lb0 = (unsigned)(uintptr_t)&Bs[0][cr][cc];
    const unsigned lb1 = (unsigned)(uintptr_t)&Bs[0][cr + 64][cc];
    const unsigned aStride = (unsigned)(TILE_M * TILE_K * 2);   // bytes per buffer
    const unsigned bStride = (unsigned)(TILE_N * TILE_K * 2);

    auto issue_tile = [&](int k0, int buf) {
        unsigned ao = (unsigned)buf * aStride;
        unsigned bo = (unsigned)buf * bStride;
        asm volatile("global_load_async_to_lds_b128 %0, %1, off"
                     :: "v"(la0 + ao),
                        "v"((unsigned long long)(uintptr_t)(ap0 + k0)) : "memory");
        asm volatile("global_load_async_to_lds_b128 %0, %1, off"
                     :: "v"(la1 + ao),
                        "v"((unsigned long long)(uintptr_t)(ap1 + k0)) : "memory");
        asm volatile("global_load_async_to_lds_b128 %0, %1, off"
                     :: "v"(lb0 + bo),
                        "v"((unsigned long long)(uintptr_t)(bp0 + k0)) : "memory");
        asm volatile("global_load_async_to_lds_b128 %0, %1, off"
                     :: "v"(lb1 + bo),
                        "v"((unsigned long long)(uintptr_t)(bp1 + k0)) : "memory");
    };

    v8f acc[2][4] = {};

    const int nk = K / TILE_K;
    issue_tile(0, 0);
    asm volatile("s_wait_asynccnt 0x0" ::: "memory");
    __syncthreads();

    for (int kt = 0; kt < nk; ++kt) {
        const int buf = kt & 1;
        if (kt + 1 < nk) issue_tile((kt + 1) * TILE_K, (kt + 1) & 1);

        // A fragments: 2 x ds_load_b128 each
        v16h af[2];
        #pragma unroll
        for (int r = 0; r < 2; ++r) {
            const v8h* arow = (const v8h*)&As[buf][wm * 32 + r * 16 + l15][0];
            v8h lo = arow[lhalf];          // K 0..15 region (half-selected)
            v8h hi = arow[2 + lhalf];      // K 16..31 region
            af[r] = __builtin_shufflevector(lo, hi,
                    0, 1, 2, 3, 4, 5, 6, 7, 8, 9, 10, 11, 12, 13, 14, 15);
        }
        #pragma unroll
        for (int j = 0; j < 4; ++j) {
            const v8h* brow = (const v8h*)&Bs[buf][wn * 64 + j * 16 + l15][0];
            v8h lo = brow[lhalf];
            v8h hi = brow[2 + lhalf];
            v16h bf = __builtin_shufflevector(lo, hi,
                      0, 1, 2, 3, 4, 5, 6, 7, 8, 9, 10, 11, 12, 13, 14, 15);
            #pragma unroll
            for (int r = 0; r < 2; ++r) {
                acc[r][j] = __builtin_amdgcn_wmma_f32_16x16x32_f16(
                    false, af[r], false, bf, (short)0, acc[r][j], false, false);
            }
        }
        // next tile must have landed (ASYNCcnt) for every wave (barrier)
        asm volatile("s_wait_asynccnt 0x0" ::: "memory");
        __syncthreads();
    }

    // epilogue: lane holds col = l15 of each 16-wide tile, rows lhalf*8 + v
    #pragma unroll
    for (int r = 0; r < 2; ++r) {
        #pragma unroll
        for (int j = 0; j < 4; ++j) {
            int col = colBase + wn * 64 + j * 16 + l15;
            #pragma unroll
            for (int v = 0; v < 8; ++v) {
                int row = rowBase + wm * 32 + r * 16 + lhalf * 8 + v;
                if (row < M) {
                    float val = acc[r][j][v];
                    if (bias)   val += bias[col];
                    if (addend) val += addend[(size_t)row * N + col];
                    if (posadd) val += posadd[(size_t)(row % NSEQ) * N + col];
                    if (relu)   val = fmaxf(val, 0.0f);
                    if (outF) outF[(size_t)row * N + col] = val;
                    if (outH) outH[(size_t)row * N + col] = (_Float16)val;
                }
            }
        }
    }
}

// ---------------------------------------------------------------------------
__global__ void f2h_kernel(const float* __restrict__ in,
                           _Float16* __restrict__ out, int n)
{
    int i = blockIdx.x * 256 + threadIdx.x;
    if (i < n) out[i] = (_Float16)in[i];
}

// value-embed layer 1: t[row][j] = relu(x[row]*w1[j] + b1[j]) -> f16
__global__ void vemb1_kernel(const float* __restrict__ x,
                             const float* __restrict__ w1,
                             const float* __restrict__ b1,
                             _Float16* __restrict__ t)
{
    int row = blockIdx.x, j = threadIdx.x;  // 256 threads
    float val = x[row] * w1[j] + b1[j];
    t[(size_t)row * 256 + j] = (_Float16)fmaxf(val, 0.0f);
}

// ---------------------------------------------------------------------------
// Attention: one block per (b,h). K tile + score matrix in LDS.
// ---------------------------------------------------------------------------
__global__ __launch_bounds__(128)
void attn_kernel(const _Float16* __restrict__ qkv,  // [B*N, 3*D]
                 _Float16* __restrict__ ao,         // [B*N, D]
                 int Dm)
{
    __shared__ _Float16 ks[NSEQ][HDIM];   // 12.8 KB
    __shared__ float    sc[NSEQ][NSEQ];   // 40 KB
    int blk = blockIdx.x;
    int b = blk >> 3, h = blk & 7;
    int tid = threadIdx.x;
    int s3 = 3 * Dm;

    for (int i = tid; i < NSEQ * HDIM; i += 128) {
        int s = i >> 6, d = i & 63;
        ks[s][d] = qkv[(size_t)(b * NSEQ + s) * s3 + Dm + h * HDIM + d];
    }
    __syncthreads();

    const float scale = 0.125f;  // 1/sqrt(64)
    for (int p = tid; p < NSEQ * NSEQ; p += 128) {
        int t = p / NSEQ, s = p % NSEQ;
        const _Float16* qp = &qkv[(size_t)(b * NSEQ + t) * s3 + h * HDIM];
        float acc = 0.f;
        #pragma unroll 8
        for (int d = 0; d < HDIM; ++d) acc += (float)qp[d] * (float)ks[s][d];
        sc[t][s] = acc * scale;
    }
    __syncthreads();

    for (int t = tid; t < NSEQ; t += 128) {
        float mx = -1e30f;
        for (int s = 0; s < NSEQ; ++s) mx = fmaxf(mx, sc[t][s]);
        float sum = 0.f;
        for (int s = 0; s < NSEQ; ++s) { float e = __expf(sc[t][s] - mx); sc[t][s] = e; sum += e; }
        float inv = 1.0f / sum;
        for (int s = 0; s < NSEQ; ++s) sc[t][s] *= inv;
    }
    __syncthreads();

    for (int p = tid; p < NSEQ * HDIM; p += 128) {
        int t = p >> 6, d = p & 63;
        float acc = 0.f;
        for (int s = 0; s < NSEQ; ++s)
            acc += sc[t][s] * (float)qkv[(size_t)(b * NSEQ + s) * s3 + 2 * Dm + h * HDIM + d];
        ao[(size_t)(b * NSEQ + t) * Dm + h * HDIM + d] = (_Float16)acc;
    }
}

// ---------------------------------------------------------------------------
__global__ __launch_bounds__(256)
void ln_kernel(const float* __restrict__ in, const float* __restrict__ g,
               const float* __restrict__ bb, float* __restrict__ outF,
               _Float16* __restrict__ outH, int D)
{
    __shared__ float red[256];
    int row = blockIdx.x, tid = threadIdx.x;
    const float* x = in + (size_t)row * D;

    float s = 0.f;
    for (int i = tid; i < D; i += 256) s += x[i];
    red[tid] = s; __syncthreads();
    for (int off = 128; off > 0; off >>= 1) { if (tid < off) red[tid] += red[tid + off]; __syncthreads(); }
    float mean = red[0] / D;
    __syncthreads();

    float v = 0.f;
    for (int i = tid; i < D; i += 256) { float d = x[i] - mean; v += d * d; }
    red[tid] = v; __syncthreads();
    for (int off = 128; off > 0; off >>= 1) { if (tid < off) red[tid] += red[tid + off]; __syncthreads(); }
    float rstd = rsqrtf(red[0] / D + 1e-5f);

    for (int i = tid; i < D; i += 256) {
        float val = (x[i] - mean) * rstd * g[i] + bb[i];
        if (outF) outF[(size_t)row * D + i] = val;
        if (outH) outH[(size_t)row * D + i] = (_Float16)val;
    }
}

// ---------------------------------------------------------------------------
// Coverage-masked argmax scan. One block per batch, 100 sequential steps.
// base: [100, B*100] with base[t*BN + b*100 + s]; out: [B,100,100]
// ---------------------------------------------------------------------------
__global__ __launch_bounds__(128)
void scan_kernel(const float* __restrict__ base, float* __restrict__ out, int Bc)
{
    __shared__ float mask[NSEQ];
    __shared__ float bv[128];
    __shared__ int   bi[128];
    int b = blockIdx.x, tid = threadIdx.x;
    if (tid < NSEQ) mask[tid] = 1.0f;
    __syncthreads();
    const size_t BN = (size_t)Bc * NSEQ;
    for (int t = 0; t < NSEQ; ++t) {
        float v = -INFINITY; int idx = 0;
        if (tid < NSEQ) {
            float lg = (mask[tid] == 0.f) ? -INFINITY
                                          : base[(size_t)t * BN + (size_t)b * NSEQ + tid];
            out[((size_t)b * NSEQ + t) * NSEQ + tid] = lg;
            v = lg; idx = tid;
        }
        bv[tid] = v; bi[tid] = idx; __syncthreads();
        for (int off = 64; off > 0; off >>= 1) {
            if (tid < off) {
                float v2 = bv[tid + off]; int i2 = bi[tid + off];
                if (v2 > bv[tid] || (v2 == bv[tid] && i2 < bi[tid])) { bv[tid] = v2; bi[tid] = i2; }
            }
            __syncthreads();
        }
        if (tid == 0) mask[bi[0]] = 0.f;
        __syncthreads();
    }
}

// ---------------------------------------------------------------------------
extern "C" void kernel_launch(void* const* d_in, const int* in_sizes, int n_in,
                              void* d_out, int out_size, void* d_ws, size_t ws_size,
                              hipStream_t stream)
{
    const float* x          = (const float*)d_in[0];
    const float* ve_w1      = (const float*)d_in[1];
    const float* ve_b1      = (const float*)d_in[2];
    const float* ve_w2      = (const float*)d_in[3];
    const float* ve_b2      = (const float*)d_in[4];
    const float* pos_emb    = (const float*)d_in[5];
    const float* attn_in_w  = (const float*)d_in[6];
    const float* attn_in_b  = (const float*)d_in[7];
    const float* attn_out_w = (const float*)d_in[8];
    const float* attn_out_b = (const float*)d_in[9];
    const float* ffn_w1     = (const float*)d_in[10];
    const float* ffn_b1     = (const float*)d_in[11];
    const float* ffn_w2     = (const float*)d_in[12];
    const float* ffn_b2     = (const float*)d_in[13];
    const float* ln1_g      = (const float*)d_in[14];
    const float* ln1_b      = (const float*)d_in[15];
    const float* ln2_g      = (const float*)d_in[16];
    const float* ln2_b      = (const float*)d_in[17];
    const float* rank_emb   = (const float*)d_in[18];
    const float* qp_w       = (const float*)d_in[19];
    const float* qp_b       = (const float*)d_in[20];
    const float* kp_w       = (const float*)d_in[21];
    const float* kp_b       = (const float*)d_in[22];

    const int Bc = 256, Dm = 512, DFFc = 2048;
    const size_t BN = (size_t)Bc * NSEQ;  // 25600

    char* ws = (char*)d_ws;
    size_t off = 0;
    auto carve = [&](size_t bytes) {
        size_t o = off; off += (bytes + 255) & ~(size_t)255; return o;
    };
    // f16 weights
    _Float16* wh_ve2  = (_Float16*)(ws + carve(2 * 512 * 256));
    _Float16* wh_qkv  = (_Float16*)(ws + carve(2 * 1536 * 512));
    _Float16* wh_ao   = (_Float16*)(ws + carve(2 * 512 * 512));
    _Float16* wh_f1   = (_Float16*)(ws + carve(2 * 2048 * 512));
    _Float16* wh_f2   = (_Float16*)(ws + carve(2 * 512 * 2048));
    _Float16* wh_qp   = (_Float16*)(ws + carve(2 * 512 * 512));
    _Float16* wh_kp   = (_Float16*)(ws + carve(2 * 512 * 512));
    _Float16* rank_h  = (_Float16*)(ws + carve(2 * NSEQ * 512));
    // activations
    _Float16* t_h     = (_Float16*)(ws + carve(2 * BN * 256));
    float*    hf      = (float*)   (ws + carve(4 * BN * Dm));   // reused as z(f32)
    _Float16* hb      = (_Float16*)(ws + carve(2 * BN * Dm));   // reused as z(f16)
    _Float16* qkv_h   = (_Float16*)(ws + carve(2 * BN * 3 * Dm));
    _Float16* ao_h    = (_Float16*)(ws + carve(2 * BN * Dm));
    float*    sum_f   = (float*)   (ws + carve(4 * BN * Dm));   // attn-proj+res, then ffn2+res
    float*    x1f     = (float*)   (ws + carve(4 * BN * Dm));
    _Float16* x1b     = (_Float16*)(ws + carve(2 * BN * Dm));   // reused as kdec f16
    _Float16* ffn_h   = (_Float16*)(ws + carve(2 * BN * DFFc));
    _Float16* qdec_h  = (_Float16*)(ws + carve(2 * NSEQ * Dm));
    float*    base_f  = (float*)   (ws + carve(4 * NSEQ * BN));
    _Float16* kdec_h  = x1b;  // reuse (x1b dead after FFN1)

    // 1) weight conversions
    auto f2h = [&](const float* src, _Float16* dst, int n) {
        f2h_kernel<<<(n + 255) / 256, 256, 0, stream>>>(src, dst, n);
    };
    f2h(ve_w2,      wh_ve2, 512 * 256);
    f2h(attn_in_w,  wh_qkv, 1536 * 512);
    f2h(attn_out_w, wh_ao,  512 * 512);
    f2h(ffn_w1,     wh_f1,  2048 * 512);
    f2h(ffn_w2,     wh_f2,  512 * 2048);
    f2h(qp_w,       wh_qp,  512 * 512);
    f2h(kp_w,       wh_kp,  512 * 512);
    f2h(rank_emb,   rank_h, NSEQ * 512);

    // 2) value-embed layer 1
    vemb1_kernel<<<(int)BN, 256, 0, stream>>>(x, ve_w1, ve_b1, t_h);

    dim3 blk(256);
    const int gyBN = (int)((BN + TILE_M - 1) / TILE_M);   // 200
    const int gy1  = (NSEQ + TILE_M - 1) / TILE_M;        // 1
    // 3) h = t @ ve_w2^T + ve_b2 + pos_emb  -> hf(f32), hb(f16)
    gemm_wmma_f16<<<dim3(512 / TILE_N, gyBN), blk, 0, stream>>>(
        t_h, wh_ve2, ve_b2, nullptr, pos_emb, hf, hb, (int)BN, 512, 256, 0);
    // 4) qkv = h @ attn_in_w^T + b
    gemm_wmma_f16<<<dim3(1536 / TILE_N, gyBN), blk, 0, stream>>>(
        hb, wh_qkv, attn_in_b, nullptr, nullptr, nullptr, qkv_h, (int)BN, 1536, 512, 0);
    // 5) attention
    attn_kernel<<<Bc * 8, 128, 0, stream>>>(qkv_h, ao_h, Dm);
    // 6) proj = ao @ attn_out_w^T + b + h(residual)
    gemm_wmma_f16<<<dim3(512 / TILE_N, gyBN), blk, 0, stream>>>(
        ao_h, wh_ao, attn_out_b, hf, nullptr, sum_f, nullptr, (int)BN, 512, 512, 0);
    // 7) LN1 -> x1f, x1b
    ln_kernel<<<(int)BN, 256, 0, stream>>>(sum_f, ln1_g, ln1_b, x1f, x1b, Dm);
    // 8) ffn1 = relu(x1 @ ffn_w1^T + b)
    gemm_wmma_f16<<<dim3(2048 / TILE_N, gyBN), blk, 0, stream>>>(
        x1b, wh_f1, ffn_b1, nullptr, nullptr, nullptr, ffn_h, (int)BN, 2048, 512, 1);
    // 9) ffn2 + residual -> sum_f
    gemm_wmma_f16<<<dim3(512 / TILE_N, gyBN), blk, 0, stream>>>(
        ffn_h, wh_f2, ffn_b2, x1f, nullptr, sum_f, nullptr, (int)BN, 512, 2048, 0);
    // 10) LN2 -> z into hf/hb (reuse)
    ln_kernel<<<(int)BN, 256, 0, stream>>>(sum_f, ln2_g, ln2_b, hf, hb, Dm);
    // 11) k_dec = z @ kp_w^T + b  (f16)
    gemm_wmma_f16<<<dim3(512 / TILE_N, gyBN), blk, 0, stream>>>(
        hb, wh_kp, kp_b, nullptr, nullptr, nullptr, kdec_h, (int)BN, 512, 512, 0);
    // 12) q_dec = rank_emb @ qp_w^T + b  (M=100, guarded)
    gemm_wmma_f16<<<dim3(512 / TILE_N, gy1), blk, 0, stream>>>(
        rank_h, wh_qp, qp_b, nullptr, nullptr, nullptr, qdec_h, NSEQ, 512, 512, 0);
    // 13) base[t, b*100+s] = q_dec[t] . k_dec[b,s]   (M=100, N=25600, K=512)
    gemm_wmma_f16<<<dim3((int)(BN / TILE_N), gy1), blk, 0, stream>>>(
        qdec_h, kdec_h, nullptr, nullptr, nullptr, base_f, nullptr, NSEQ, (int)BN, 512, 0);
    // 14) masked argmax scan -> d_out [B,100,100]
    scan_kernel<<<Bc, 128, 0, stream>>>(base_f, (float*)d_out, Bc);

    (void)in_sizes; (void)n_in; (void)out_size; (void)ws_size;
}
